// MultiHeadAttention_88210038326176
// MI455X (gfx1250) — matrix operand order repaired and checked by reading, not compiled, not run on hardware
//
#include <hip/hip_runtime.h>
#include <math.h>

typedef unsigned short u16;
typedef __attribute__((ext_vector_type(16))) __bf16 v16bf;
typedef __attribute__((ext_vector_type(8)))  float  v8f;
typedef __attribute__((ext_vector_type(4)))  unsigned int u32x4;
typedef __attribute__((ext_vector_type(4)))  int v4i;

union Frag {           // 16 bf16 = 32 bytes, raw-bit moved as two 16B chunks
    v16bf v;
    u32x4 q[2];
};

// CDNA5 async global->LDS staging (ASYNCcnt path), with sync fallback.
// Probe-learned signature: (v4i AS1*, v4i AS3*, imm int offset, imm int cpol).
#if __has_builtin(__builtin_amdgcn_global_load_async_to_lds_b128) && \
    __has_builtin(__builtin_amdgcn_s_wait_asynccnt)
#define USE_ASYNC_LDS 1
#define AS1V4(p) ((__attribute__((address_space(1))) v4i*)(p))
#define AS3V4(p) ((__attribute__((address_space(3))) v4i*)(p))
#define ASYNC_B128(gsrc, ldst) \
    __builtin_amdgcn_global_load_async_to_lds_b128( \
        AS1V4(const_cast<u16*>(gsrc)), AS3V4(ldst), 0, 0)
#else
#define USE_ASYNC_LDS 0
#endif

static __device__ inline u16 f2bf(float f) {
    unsigned int u = __float_as_uint(f);
    u += 0x7FFFu + ((u >> 16) & 1u);   // round-to-nearest-even
    return (u16)(u >> 16);
}

static __device__ inline v8f wmma_bf16(const v16bf& a, const v16bf& b, v8f c) {
    return __builtin_amdgcn_wmma_f32_16x16x32_bf16(
        /*neg_a=*/false, a, /*neg_b=*/false, b,
        /*c_mod=*/(short)0, c, /*reuse_a=*/false, /*reuse_b=*/false);
}

static __device__ inline float halfmax(float v) {
    #pragma unroll
    for (int m = 1; m < 16; m <<= 1) v = fmaxf(v, __shfl_xor(v, m, 32));
    return v;   // reduced across the 16-lane half (masks<16 stay in-half)
}
static __device__ inline float halfsum(float v) {
    #pragma unroll
    for (int m = 1; m < 16; m <<= 1) v += __shfl_xor(v, m, 32);
    return v;
}

// ---------------- constants ----------------
#define BATCH 8
#define TOK   1024
#define CH    384
#define NH    12
#define DH    32
#define MTOT  (BATCH * TOK)    // 8192

// ---------------- kernel 1: depthwise conv 3x3 + BN -> y (bf16) ----------------
__global__ __launch_bounds__(256) void convbn_kernel(
    const float* __restrict__ x, const float* __restrict__ cw,
    const float* __restrict__ g, const float* __restrict__ be,
    const float* __restrict__ mu, const float* __restrict__ var,
    u16* __restrict__ yb) {
    int idx = blockIdx.x * 256 + threadIdx.x;
    if (idx >= BATCH * TOK * CH) return;
    int c = idx % CH;
    int t = (idx / CH) % TOK;
    int b = idx / (CH * TOK);
    int hh = t >> 5, ww = t & 31;
    float acc = 0.f;
    #pragma unroll
    for (int dh = -1; dh <= 1; ++dh) {
        #pragma unroll
        for (int dw = -1; dw <= 1; ++dw) {
            int h2 = hh + dh, w2 = ww + dw;
            if (h2 >= 0 && h2 < 32 && w2 >= 0 && w2 < 32) {
                acc += x[((size_t)b * TOK + (h2 << 5) + w2) * CH + c]
                     * cw[c * 9 + (dh + 1) * 3 + (dw + 1)];
            }
        }
    }
    float inv = g[c] * rsqrtf(var[c] + 1e-5f);
    yb[idx] = f2bf(acc * inv + (be[c] - mu[c] * inv));
}

// ---------------- kernel 2: f32 -> bf16 ----------------
__global__ __launch_bounds__(256) void cvt_kernel(const float* __restrict__ in,
                                                  u16* __restrict__ out, int n) {
    int i = blockIdx.x * 256 + threadIdx.x;
    if (i < n) out[i] = f2bf(in[i]);
}

// ---------------- kernel 3: GEMM (M=8192, N=384, K=384), Y @ W^T ----------------
// A: y bf16 row-major [M][K].  W bf16 row-major [N][K] (torch Linear layout) ==
// exactly the (n,k) tile layout the B-fragment wants.
// mode 0: store f32 row-major to outF.
// mode 1: store bf16 * scale into [b,head,t,d] head-major layout (outH).
__global__ __launch_bounds__(256) void gemm_kernel(
    const u16* __restrict__ A, const u16* __restrict__ W,
    float* __restrict__ outF, u16* __restrict__ outH, int mode, float scale) {
    __shared__ __align__(16) u16 As[128 * 32];
    __shared__ __align__(16) u16 Bs[64 * 32];

    const int tid  = threadIdx.x;
    const int lane = tid & 31;
    const int wave = tid >> 5;
    const int wm = wave >> 1, wn = wave & 1;          // 4x2 waves of 32x32 tiles
    const int mBase = blockIdx.x * 128;
    const int nBase = blockIdx.y * 64;
    const int kb  = (lane >> 4) * 8;
    const int kb2 = (lane >> 4) * 16;

    v8f acc[2][2];
    #pragma unroll
    for (int i = 0; i < 2; ++i)
        #pragma unroll
        for (int j = 0; j < 2; ++j) acc[i][j] = (v8f)0.f;

    for (int k0 = 0; k0 < CH; k0 += 32) {
        __syncthreads();
        {   // stage A tile 128x32 (16 bf16 per thread, two b128)
            int i = tid * 16;
            int row = i >> 5, col = i & 31;
            const u16* src = A + (size_t)(mBase + row) * CH + k0 + col;
#if USE_ASYNC_LDS
            ASYNC_B128(src, &As[i]);
            ASYNC_B128(src + 8, &As[i + 8]);
#else
            *(u32x4*)&As[i]     = *(const u32x4*)src;
            *(u32x4*)&As[i + 8] = *(const u32x4*)(src + 8);
#endif
            if (k0 + 32 < CH) __builtin_prefetch(src + 32, 0, 0);
        }
        {   // stage B tile 64x32 (8 bf16 per thread)
            int i = tid * 8;
            int row = i >> 5, col = i & 31;
            const u16* src = W + (size_t)(nBase + row) * CH + k0 + col;
#if USE_ASYNC_LDS
            ASYNC_B128(src, &Bs[i]);
#else
            *(u32x4*)&Bs[i] = *(const u32x4*)src;
#endif
            if (k0 + 32 < CH) __builtin_prefetch(src + 32, 0, 0);
        }
#if USE_ASYNC_LDS
        __builtin_amdgcn_s_wait_asynccnt(0);
#endif
        __syncthreads();

        Frag am[2], bn[2];
        #pragma unroll
        for (int mt = 0; mt < 2; ++mt) {
            int m = wm * 32 + mt * 16 + (lane & 15);
            am[mt].q[0] = *(u32x4*)&As[m * 32 + kb];
            am[mt].q[1] = *(u32x4*)&As[m * 32 + kb + 16];
        }
        #pragma unroll
        for (int nt = 0; nt < 2; ++nt) {
            int n = wn * 32 + nt * 16 + (lane & 15);
            bn[nt].q[0] = *(u32x4*)&Bs[n * 32 + kb];
            bn[nt].q[1] = *(u32x4*)&Bs[n * 32 + kb + 16];
        }
        #pragma unroll
        for (int mt = 0; mt < 2; ++mt)
            #pragma unroll
            for (int nt = 0; nt < 2; ++nt)
                acc[mt][nt] = wmma_bf16(am[mt].v, bn[nt].v, acc[mt][nt]);
    }

    // epilogue: C/D layout rows m = r + 8*(lane>=16), col n = lane&15
    const int h = lane >> 4;
    #pragma unroll
    for (int mt = 0; mt < 2; ++mt) {
        #pragma unroll
        for (int nt = 0; nt < 2; ++nt) {
            int nG = nBase + wn * 32 + nt * 16 + (lane & 15);
            #pragma unroll
            for (int r = 0; r < 8; ++r) {
                int mG = mBase + wm * 32 + mt * 16 + r + 8 * h;
                float v = acc[mt][nt][r];
                if (mode == 0) {
                    outF[(size_t)mG * CH + nG] = v;
                } else {
                    int head = nG >> 5, d = nG & 31;
                    int b = mG >> 10, t = mG & 1023;
                    outH[(((size_t)(b * NH + head)) * TOK + t) * DH + d] =
                        f2bf(v * scale);
                }
            }
        }
    }
}

// ---------------- kernel 4: flash attention, one WG per (b*h, 128-query block) ----
// Q already carries the sqrt(d) scale. Q/K/V in [bh][t][d] bf16.
__global__ __launch_bounds__(256) void attn_kernel(
    const u16* __restrict__ Qh, const u16* __restrict__ Kh,
    const u16* __restrict__ Vh, u16* __restrict__ Ob) {
    __shared__ __align__(16) u16 Ks[128 * 32];      // [key][d]  == (n,k) for S
    __shared__ __align__(16) u16 Vt[32 * 128];      // [d][key]  == (n,k) for O
    __shared__ __align__(16) u16 Ps[8 * 16 * 32];   // per-wave P scratch

    const int tid  = threadIdx.x;
    const int lane = tid & 31;
    const int wave = tid >> 5;
    const int bh = blockIdx.y;
    const int b = bh / NH, head = bh - b * NH;
    const int qRow = blockIdx.x * 128 + wave * 16;
    const int kb  = (lane >> 4) * 8;
    const int kb2 = (lane >> 4) * 16;
    const int n = lane & 15;
    const int h = lane >> 4;

    const u16* Q = Qh + (size_t)bh * TOK * DH;
    const u16* K = Kh + (size_t)bh * TOK * DH;
    const u16* V = Vh + (size_t)bh * TOK * DH;

    // Q fragment for this wave's 16 rows (held the whole kernel)
    Frag aQ;
    aQ.q[0] = *(const u32x4*)(Q + (size_t)(qRow + n) * DH + kb);
    aQ.q[1] = *(const u32x4*)(Q + (size_t)(qRow + n) * DH + kb + 16);

    v8f o0 = (v8f)0.f, o1 = (v8f)0.f;
    float mrun[8], lrun[8];
    #pragma unroll
    for (int r = 0; r < 8; ++r) { mrun[r] = -1e30f; lrun[r] = 0.f; }

    u16* Pw = Ps + wave * 512;

    for (int kc = 0; kc < TOK; kc += 128) {
        __syncthreads();
        {   // K chunk, straight copy (layout already (key,d) row-major)
            int i = tid * 16;
            const u16* src = K + (size_t)kc * DH + i;
#if USE_ASYNC_LDS
            ASYNC_B128(src, &Ks[i]);
            ASYNC_B128(src + 8, &Ks[i + 8]);
#else
            *(u32x4*)&Ks[i]     = *(const u32x4*)src;
            *(u32x4*)&Ks[i + 8] = *(const u32x4*)(src + 8);
#endif
            if (kc + 128 < TOK) __builtin_prefetch(src + 128 * DH, 0, 0);
        }
        {   // V chunk, transposed into Vt[d][key]
            #pragma unroll
            for (int e = 0; e < 16; ++e) {
                int i = tid + e * 256;
                int key = i >> 5, d = i & 31;
                Vt[d * 128 + key] = V[(size_t)(kc + key) * DH + d];
            }
        }
#if USE_ASYNC_LDS
        __builtin_amdgcn_s_wait_asynccnt(0);
#endif
        __syncthreads();

        for (int jt = 0; jt < 4; ++jt) {            // 32 keys per step
            Frag bk0, bk1;
            bk0.q[0] = *(u32x4*)&Ks[(jt * 32 + n) * 32 + kb];
            bk0.q[1] = *(u32x4*)&Ks[(jt * 32 + n) * 32 + kb + 16];
            bk1.q[0] = *(u32x4*)&Ks[(jt * 32 + 16 + n) * 32 + kb];
            bk1.q[1] = *(u32x4*)&Ks[(jt * 32 + 16 + n) * 32 + kb + 16];

            v8f s0 = wmma_bf16(aQ.v, bk0.v, (v8f)0.f);
            v8f s1 = wmma_bf16(aQ.v, bk1.v, (v8f)0.f);

            // online softmax over this 32-key block
            #pragma unroll
            for (int r = 0; r < 8; ++r) {
                float tm   = halfmax(fmaxf(s0[r], s1[r]));
                float mnew = fmaxf(mrun[r], tm);
                float corr = __expf(mrun[r] - mnew);
                float p0   = __expf(s0[r] - mnew);
                float p1   = __expf(s1[r] - mnew);
                float ts   = halfsum(p0 + p1);
                lrun[r] = lrun[r] * corr + ts;
                mrun[r] = mnew;
                o0[r] *= corr;
                o1[r] *= corr;
                int row = r + 8 * h;
                Pw[row * 32 + n]      = f2bf(p0);
                Pw[row * 32 + 16 + n] = f2bf(p1);
            }

            // re-fetch P as an A-fragment (C-layout -> A-layout via LDS)
            Frag aP;
            aP.q[0] = *(u32x4*)&Pw[n * 32 + kb];
            aP.q[1] = *(u32x4*)&Pw[n * 32 + kb + 16];

            Frag bv0, bv1;
            bv0.q[0] = *(u32x4*)&Vt[n * 128 + jt * 32 + kb];
            bv0.q[1] = *(u32x4*)&Vt[n * 128 + jt * 32 + kb + 16];
            bv1.q[0] = *(u32x4*)&Vt[(16 + n) * 128 + jt * 32 + kb];
            bv1.q[1] = *(u32x4*)&Vt[(16 + n) * 128 + jt * 32 + kb + 16];

            o0 = wmma_bf16(aP.v, bv0.v, o0);
            o1 = wmma_bf16(aP.v, bv1.v, o1);
        }
    }

    // normalize + store to [b, t, head*32+d] bf16 (feeds output projection)
    #pragma unroll
    for (int r = 0; r < 8; ++r) {
        float inv = 1.0f / lrun[r];
        int t = qRow + r + 8 * h;
        size_t base = ((size_t)(b * TOK) + t) * CH + head * DH;
        Ob[base + n]      = f2bf(o0[r] * inv);
        Ob[base + 16 + n] = f2bf(o1[r] * inv);
    }
}

// ---------------- host launcher ----------------
extern "C" void kernel_launch(void* const* d_in, const int* in_sizes, int n_in,
                              void* d_out, int out_size, void* d_ws, size_t ws_size,
                              hipStream_t stream) {
    const float* x    = (const float*)d_in[0];
    const float* cw   = (const float*)d_in[1];
    const float* g    = (const float*)d_in[2];
    const float* be   = (const float*)d_in[3];
    const float* mu   = (const float*)d_in[4];
    const float* var  = (const float*)d_in[5];
    const float* wq   = (const float*)d_in[6];
    const float* wk   = (const float*)d_in[7];
    const float* wv   = (const float*)d_in[8];
    const float* wo   = (const float*)d_in[9];
    (void)in_sizes; (void)n_in; (void)out_size; (void)ws_size;

    char* ws = (char*)d_ws;
    const size_t Ybytes = (size_t)MTOT * CH * sizeof(u16);   // 6.29 MB
    const size_t Wbytes = (size_t)CH * CH * sizeof(u16);     // 0.29 MB

    u16* yb   = (u16*)ws;              ws += Ybytes;
    u16* wqb  = (u16*)ws;              ws += Wbytes;
    u16* wkb  = (u16*)ws;              ws += Wbytes;
    u16* wvb  = (u16*)ws;              ws += Wbytes;
    u16* wob  = (u16*)ws;              ws += Wbytes;
    u16* Qh   = (u16*)ws;              ws += Ybytes;
    u16* Kh   = (u16*)ws;              ws += Ybytes;
    u16* Vh   = (u16*)ws;              ws += Ybytes;
    u16* atb  = (u16*)ws;              ws += Ybytes;

    // 1) conv + BN -> y bf16
    convbn_kernel<<<(BATCH * TOK * CH) / 256, 256, 0, stream>>>(x, cw, g, be, mu, var, yb);

    // 2) weights -> bf16
    const int wn = CH * CH;
    cvt_kernel<<<(wn + 255) / 256, 256, 0, stream>>>(wq, wqb, wn);
    cvt_kernel<<<(wn + 255) / 256, 256, 0, stream>>>(wk, wkb, wn);
    cvt_kernel<<<(wn + 255) / 256, 256, 0, stream>>>(wv, wvb, wn);
    cvt_kernel<<<(wn + 255) / 256, 256, 0, stream>>>(wo, wob, wn);

    // 3) Q/K/V projections into head-major bf16; fold sqrt(d_head) into Q
    dim3 ggrid(MTOT / 128, CH / 64);
    const float sqrt_d = 5.656854249492381f;  // sqrt(32), reference MULTIPLIES
    gemm_kernel<<<ggrid, 256, 0, stream>>>(yb, wqb, nullptr, Qh, 1, sqrt_d);
    gemm_kernel<<<ggrid, 256, 0, stream>>>(yb, wkb, nullptr, Kh, 1, 1.0f);
    gemm_kernel<<<ggrid, 256, 0, stream>>>(yb, wvb, nullptr, Vh, 1, 1.0f);

    // 4) attention
    attn_kernel<<<dim3(TOK / 128, BATCH * NH), 256, 0, stream>>>(Qh, Kh, Vh, atb);

    // 5) output projection -> f32 d_out
    gemm_kernel<<<ggrid, 256, 0, stream>>>(atb, wob, (float*)d_out, nullptr, 0, 1.0f);
}
